// BFPLinear_32916629357323
// MI455X (gfx1250) — matrix operand order repaired and checked by reading, compiled, not run on hardware
//
#include <hip/hip_runtime.h>
#include <stdint.h>

typedef unsigned short u16;
typedef __attribute__((ext_vector_type(16))) __bf16 v16bf;
typedef __attribute__((ext_vector_type(8)))  float  v8f;

union Frag { uint4 u4[2]; v16bf v; };

#define M_DIM 8192
#define N_DIM 4096
#define K_DIM 4096
#define BM 128
#define BN 128
#define BK 64

#if __has_builtin(__builtin_amdgcn_global_load_async_to_lds_b128)
#define HAVE_ASYNC_LDS 1
#else
#define HAVE_ASYNC_LDS 0
#endif

#if HAVE_ASYNC_LDS
typedef __attribute__((__vector_size__(4 * sizeof(int)))) int v4i_t;
typedef __attribute__((address_space(1))) v4i_t* g_v4i_p;
typedef __attribute__((address_space(3))) v4i_t* l_v4i_p;

__device__ __forceinline__ void async_g2l_b128(const u16* g, u16* l) {
  __builtin_amdgcn_global_load_async_to_lds_b128((g_v4i_p)g, (l_v4i_p)l, 0, 0);
}
__device__ __forceinline__ void wait_async0() {
#if __has_builtin(__builtin_amdgcn_s_wait_asynccnt)
  __builtin_amdgcn_s_wait_asynccnt(0);
#else
  asm volatile("s_wait_asynccnt 0x0" ::: "memory");
#endif
}
#endif

// ---------------------------------------------------------------------------
// BFP quantize (GROUP=32, BITS=8) -> bf16 (exact: |q|<=127 times power of 2)
// ---------------------------------------------------------------------------
__device__ __forceinline__ u16 f32_to_bf16_bits(float f) {
  return (u16)(__float_as_uint(f) >> 16);  // exact: value fits in 8-bit significand
}

__global__ __launch_bounds__(256) void bfp_quant_bf16(
    const float* __restrict__ in, u16* __restrict__ out, long n)
{
  long base = ((long)blockIdx.x * 256 + (long)threadIdx.x) * 4;
  if (base + 4 > n) return;
  float4 v = *(const float4*)(in + base);

  float m = fmaxf(fmaxf(fabsf(v.x), fabsf(v.y)), fmaxf(fabsf(v.z), fabsf(v.w)));
  m = fmaxf(m, __shfl_xor(m, 1, 32));
  m = fmaxf(m, __shfl_xor(m, 2, 32));
  m = fmaxf(m, __shfl_xor(m, 4, 32));

  int e = (int)((__float_as_uint(m) >> 23) & 0xFF) - 127;
  float stepv = ldexpf(1.0f, e - 6);
  float inv   = ldexpf(1.0f, 6 - e);

  float q0 = fminf(fmaxf(rintf(v.x * inv), -127.0f), 127.0f) * stepv;
  float q1 = fminf(fmaxf(rintf(v.y * inv), -127.0f), 127.0f) * stepv;
  float q2 = fminf(fmaxf(rintf(v.z * inv), -127.0f), 127.0f) * stepv;
  float q3 = fminf(fmaxf(rintf(v.w * inv), -127.0f), 127.0f) * stepv;
  if (!(m > 0.0f)) { q0 = q1 = q2 = q3 = 0.0f; }

  ushort4 o;
  o.x = f32_to_bf16_bits(q0);
  o.y = f32_to_bf16_bits(q1);
  o.z = f32_to_bf16_bits(q2);
  o.w = f32_to_bf16_bits(q3);
  *(ushort4*)(out + base) = o;
}

// ---------------------------------------------------------------------------
// GEMM: out[m,n] = sum_k Aq[m,k]*Bq[n,k] + 2*bias[n]
// 128x128 tile, K-tile 64, 8 waves (4xM, 2xN), 2x4 WMMA accs per wave.
// Double-buffered LDS fed by ASYNC global->LDS copies (no VGPR staging).
// ---------------------------------------------------------------------------
__global__ __launch_bounds__(256, 1) void bfp_gemm_bf16(
    const u16* __restrict__ Aq, const u16* __restrict__ Bq,
    const float* __restrict__ bias, float* __restrict__ out)
{
  __shared__ u16 As[2][BM][BK];   // 32 KB
  __shared__ u16 Bs[2][BN][BK];   // 32 KB

  const int tid  = threadIdx.x;
  const int lane = tid & 31;
  const int wave = tid >> 5;
  const int wm   = wave & 3;
  const int wn   = wave >> 2;
  const int m0   = blockIdx.y * BM;
  const int n0   = blockIdx.x * BN;

  int rowT[4], colT[4];
#pragma unroll
  for (int i = 0; i < 4; ++i) {
    int c = tid + i * 256;          // 1024 x 16B chunks per 128x64 bf16 tile
    rowT[i] = c >> 3;
    colT[i] = (c & 7) * 8;
  }

  const u16* gA[4];
  const u16* gB[4];
#pragma unroll
  for (int i = 0; i < 4; ++i) {
    gA[i] = Aq + (size_t)(m0 + rowT[i]) * K_DIM + colT[i];
    gB[i] = Bq + (size_t)(n0 + rowT[i]) * K_DIM + colT[i];
  }

#if HAVE_ASYNC_LDS
  auto prefetch = [&](int kt, int buf) {
    const int k0 = kt * BK;
#pragma unroll
    for (int i = 0; i < 4; ++i) {
      async_g2l_b128(gA[i] + k0, &As[buf][rowT[i]][colT[i]]);
      async_g2l_b128(gB[i] + k0, &Bs[buf][rowT[i]][colT[i]]);
    }
  };
#else
  uint4 ra[4], rb[4];
  auto loadG = [&](int kt) {
    const int k0 = kt * BK;
#pragma unroll
    for (int i = 0; i < 4; ++i) {
      ra[i] = *(const uint4*)(gA[i] + k0);
      rb[i] = *(const uint4*)(gB[i] + k0);
    }
  };
  auto storeL = [&](int buf) {
#pragma unroll
    for (int i = 0; i < 4; ++i) {
      *(uint4*)&As[buf][rowT[i]][colT[i]] = ra[i];
      *(uint4*)&Bs[buf][rowT[i]][colT[i]] = rb[i];
    }
  };
#endif

  v8f acc[2][4];
#pragma unroll
  for (int i = 0; i < 2; ++i)
#pragma unroll
    for (int j = 0; j < 4; ++j)
      acc[i][j] = {};

#if HAVE_ASYNC_LDS
  prefetch(0, 0);
  wait_async0();
#else
  loadG(0);
  storeL(0);
#endif

  const int KT = K_DIM / BK;  // 64
  for (int kt = 0; kt < KT; ++kt) {
    __syncthreads();
    const bool more = (kt + 1) < KT;

#if HAVE_ASYNC_LDS
    if (more) prefetch(kt + 1, (kt + 1) & 1);
#else
    if (more) loadG(kt + 1);
#endif

    const int buf = kt & 1;
#pragma unroll
    for (int ks = 0; ks < 2; ++ks) {
      const int koff = ks * 32;

      Frag a[2];
      const int ahalf = (lane < 16) ? 0 : 16;
#pragma unroll
      for (int i = 0; i < 2; ++i) {
        const u16* ap = &As[buf][wm * 32 + i * 16 + (lane & 15)][koff + ahalf];
        a[i].u4[0] = *(const uint4*)ap;
        a[i].u4[1] = *(const uint4*)(ap + 8);
      }

      Frag b[4];
      const int bhalf = (lane < 16) ? 0 : 16;
#pragma unroll
      for (int j = 0; j < 4; ++j) {
        const u16* bp = &Bs[buf][wn * 64 + j * 16 + (lane & 15)][koff + bhalf];
        b[j].u4[0] = *(const uint4*)bp;
        b[j].u4[1] = *(const uint4*)(bp + 8);
      }

#pragma unroll
      for (int i = 0; i < 2; ++i)
#pragma unroll
        for (int j = 0; j < 4; ++j)
          acc[i][j] = __builtin_amdgcn_wmma_f32_16x16x32_bf16(
              false, a[i].v, false, b[j].v, (short)0, acc[i][j], false, false);
    }

#if HAVE_ASYNC_LDS
    if (more) wait_async0();
#else
    if (more) storeL((kt + 1) & 1);
#endif
  }

#pragma unroll
  for (int j = 0; j < 4; ++j) {
    const int col = n0 + wn * 64 + j * 16 + (lane & 15);
    const float bb = 2.0f * bias[col];
#pragma unroll
    for (int i = 0; i < 2; ++i) {
      const int mbase = m0 + wm * 32 + i * 16 + ((lane < 16) ? 0 : 8);
#pragma unroll
      for (int r = 0; r < 8; ++r) {
        out[(size_t)(mbase + r) * N_DIM + col] = acc[i][j][r] + bb;
      }
    }
  }
}

// ---------------------------------------------------------------------------
extern "C" void kernel_launch(void* const* d_in, const int* in_sizes, int n_in,
                              void* d_out, int out_size, void* d_ws, size_t ws_size,
                              hipStream_t stream) {
  const float* x    = (const float*)d_in[0];
  const float* w    = (const float*)d_in[1];
  const float* bias = (const float*)d_in[2];
  float* out = (float*)d_out;

  u16* xq = (u16*)d_ws;
  u16* wq = xq + (size_t)M_DIM * K_DIM;

  const long nx = (long)M_DIM * K_DIM;
  const long nw = (long)N_DIM * K_DIM;

  bfp_quant_bf16<<<(int)(nx / 1024), 256, 0, stream>>>(x, xq, nx);
  bfp_quant_bf16<<<(int)(nw / 1024), 256, 0, stream>>>(w, wq, nw);

  dim3 grid(N_DIM / BN, M_DIM / BM);
  bfp_gemm_bf16<<<grid, 256, 0, stream>>>(xq, wq, bias, out);
}
